// PrefillChunk2_51891794870756
// MI455X (gfx1250) — compile-verified, matrix operand order checked by reading, not run on hardware
//
#include <hip/hip_runtime.h>
#include <cstdint>
#include <cstddef>

// ---------------- problem constants (match reference) ----------------
#define N_TOK   1024
#define DMODEL  2048
#define NHEAD   8
#define NKV     2
#define HDIM    256
#define FFDIM   4096
#define HPLDIM  256
#define NLAYER  7
#define TOTLAYERS 30
#define STARTL  8
#define EPSV    1e-6f

// ---------------- CDNA5 types ----------------
typedef __attribute__((ext_vector_type(16))) __bf16  v16bf;
typedef __attribute__((ext_vector_type(8)))  float   v8f;
typedef __attribute__((ext_vector_type(4)))  unsigned int u32x4_t;
typedef __attribute__((ext_vector_type(8)))  int     i32x8_t;
typedef __attribute__((ext_vector_type(4)))  int     i32x4_t;

#if defined(__HIP_DEVICE_COMPILE__) && __has_builtin(__builtin_amdgcn_tensor_load_to_lds)
#define HAVE_TDM 1
#else
#define HAVE_TDM 0
#endif

__device__ __forceinline__ void wait_tensorcnt0() {
#if defined(__HIP_DEVICE_COMPILE__)
#if __has_builtin(__builtin_amdgcn_s_wait_tensorcnt)
  __builtin_amdgcn_s_wait_tensorcnt(0);
#elif HAVE_TDM
  asm volatile("s_wait_tensorcnt 0x0" ::: "memory");
#endif
#endif
}

__device__ __forceinline__ unsigned short f2bf(float x) {
  union { __bf16 h; unsigned short u; } c;
  c.h = (__bf16)x;
  return c.u;
}

__device__ __forceinline__ float gelu_tanh(float x) {
  float x3 = x * x * x;
  return 0.5f * x * (1.0f + tanhf(0.7978845608028654f * (x + 0.044715f * x3)));
}

// =====================================================================
// Generic WMMA GEMM:  C[M x Nc] = A[M x K](bf16) * W[Nc x K]^T
//   W fp32 (converted to bf16 during LDS staging) or bf16 (TDM staged).
//   Optional additive mask (attention scores epilogue), fp32 or bf16 C.
// Block tile 128x128, 8 waves (2x4), wave tile 64x32, K-step 32.
// =====================================================================
#define BT_M 128
#define BT_N 128
#define BT_K 32

__global__ __launch_bounds__(256)
void gemm_wmma(const unsigned short* __restrict__ A, int lda,
               const void* __restrict__ Wp, int w_bf16, int ldw,
               void* __restrict__ Cp, int c_bf16, int ldc,
               const float* __restrict__ addmask, int mask_ld,
               int K, int Nc_total)
{
  __shared__ __align__(16) unsigned short sA[2][BT_M * BT_K];
  __shared__ __align__(16) unsigned short sB[2][BT_N * BT_K];

  const int t    = threadIdx.x;
  const int lane = t & 31;
  const int wid  = t >> 5;
  const int wm   = wid >> 2;      // 0..1 -> M quadrant
  const int wn   = wid & 3;       // 0..3 -> N quadrant
  const int m0   = blockIdx.y * BT_M;
  const int n0   = blockIdx.x * BT_N;
  const int nk   = K / BT_K;

  v8f acc[4][2];
  for (int i = 0; i < 4; ++i)
    for (int j = 0; j < 2; ++j)
      for (int e = 0; e < 8; ++e) acc[i][j][e] = 0.0f;

  // ---- staging: A tile (bf16, 128x32), 256 threads x 16 halves ----
  auto stageA = [&](int kt, int buf) {
    int row = t >> 1;
    int c0  = (t & 1) * 16;
    const uint4* src = (const uint4*)(A + (size_t)(m0 + row) * lda + (size_t)kt * BT_K + c0);
    uint4 x0 = src[0], x1 = src[1];
    *(uint4*)&sA[buf][row * BT_K + c0]     = x0;
    *(uint4*)&sA[buf][row * BT_K + c0 + 8] = x1;
  };

  // ---- staging: W fp32 -> bf16 (128x32) ----
  auto stageB_f32 = [&](int kt, int buf) {
    const float* W = (const float*)Wp;
    int row = t >> 1;
    int c0  = (t & 1) * 16;
    const float4* src = (const float4*)(W + (size_t)(n0 + row) * ldw + (size_t)kt * BT_K + c0);
    __builtin_prefetch(src + 16, 0, 1);   // global_prefetch next tiles
    float4 f0 = src[0], f1 = src[1], f2 = src[2], f3 = src[3];
    union { unsigned short s[8]; uint4 u; } p0, p1;
    p0.s[0] = f2bf(f0.x); p0.s[1] = f2bf(f0.y); p0.s[2] = f2bf(f0.z); p0.s[3] = f2bf(f0.w);
    p0.s[4] = f2bf(f1.x); p0.s[5] = f2bf(f1.y); p0.s[6] = f2bf(f1.z); p0.s[7] = f2bf(f1.w);
    p1.s[0] = f2bf(f2.x); p1.s[1] = f2bf(f2.y); p1.s[2] = f2bf(f2.z); p1.s[3] = f2bf(f2.w);
    p1.s[4] = f2bf(f3.x); p1.s[5] = f2bf(f3.y); p1.s[6] = f2bf(f3.z); p1.s[7] = f2bf(f3.w);
    *(uint4*)&sB[buf][row * BT_K + c0]     = p0.u;
    *(uint4*)&sB[buf][row * BT_K + c0 + 8] = p1.u;
  };

  // ---- staging: W bf16 via plain loads (fallback path) ----
  auto stageB_bf16 = [&](int kt, int buf) {
    const unsigned short* W = (const unsigned short*)Wp;
    int row = t >> 1;
    int c0  = (t & 1) * 16;
    const uint4* src = (const uint4*)(W + (size_t)(n0 + row) * ldw + (size_t)kt * BT_K + c0);
    uint4 x0 = src[0], x1 = src[1];
    *(uint4*)&sB[buf][row * BT_K + c0]     = x0;
    *(uint4*)&sB[buf][row * BT_K + c0 + 8] = x1;
  };

#if HAVE_TDM
  // ---- staging: W bf16 via Tensor Data Mover (2D tile descriptor) ----
  auto stageB_tdm = [&](int kt, int buf) {
    if (t < 32) {   // wave 0 issues the DMA (TDM ignores EXEC; wave-uniform branch)
      const unsigned short* W = (const unsigned short*)Wp;
      unsigned long long ga =
          (unsigned long long)(uintptr_t)(W + (size_t)n0 * ldw + (size_t)kt * BT_K);
      unsigned int ldsa = (unsigned int)(uintptr_t)(&sB[buf][0]);
      u32x4_t g0;
      g0[0] = 1u;                                   // count=1, user descriptor
      g0[1] = ldsa;                                 // lds_addr
      g0[2] = (unsigned int)ga;                     // global_addr[31:0]
      g0[3] = (unsigned int)(ga >> 32) | (2u << 30);// global_addr[56:32] | type=2
      i32x8_t g1;
      g1[0] = (int)(1u << 16);                      // data_size = 1 (2 bytes)
      g1[1] = (int)(((unsigned)ldw & 0xFFFFu) << 16);               // tensor_dim0 lo
      g1[2] = (int)(((unsigned)ldw >> 16) |
                    (((unsigned)Nc_total & 0xFFFFu) << 16));        // dim0 hi | dim1 lo
      g1[3] = (int)(((unsigned)Nc_total >> 16) | ((unsigned)BT_K << 16)); // dim1 hi | tile_dim0=32
      g1[4] = (int)BT_N;                            // tile_dim1=128 (tile_dim2=0)
      g1[5] = (int)(unsigned)ldw;                   // tensor_dim0_stride lo
      g1[6] = 0;                                    // stride hi | dim1_stride lo
      g1[7] = 0;
      i32x4_t gz;  gz[0] = gz[1] = gz[2] = gz[3] = 0;
#if defined(__clang_major__) && (__clang_major__ >= 23)
      i32x8_t gz8; for (int z = 0; z < 8; ++z) gz8[z] = 0;
      __builtin_amdgcn_tensor_load_to_lds(g0, g1, gz, gz, gz8, 0);
#else
      __builtin_amdgcn_tensor_load_to_lds(g0, g1, gz, gz, 0);
#endif
    }
  };
#endif

  // ---- fragment load per documented 16-bit A/B lane layout ----
  auto ldfrag = [&](const unsigned short* s, int rowBase) -> v16bf {
    int r  = rowBase + (lane & 15);
    int kb = (lane >> 4) * 8;
    union { uint4 u[2]; v16bf v; } f;
    f.u[0] = *(const uint4*)&s[r * BT_K + kb];
    f.u[1] = *(const uint4*)&s[r * BT_K + kb + 16];
    return f.v;
  };

  bool tdm_used = false;
  // prologue: stage tile 0 into buffer 0
  stageA(0, 0);
#if HAVE_TDM
  if (w_bf16) { stageB_tdm(0, 0); tdm_used = true; }
  else        { stageB_f32(0, 0); }
#else
  if (w_bf16) stageB_bf16(0, 0); else stageB_f32(0, 0);
#endif
  if (tdm_used && t < 32) wait_tensorcnt0();
  __syncthreads();

  for (int kt = 0; kt < nk; ++kt) {
    const int cur = kt & 1;
    if (kt + 1 < nk) {
      stageA(kt + 1, cur ^ 1);
#if HAVE_TDM
      if (w_bf16) stageB_tdm(kt + 1, cur ^ 1); else stageB_f32(kt + 1, cur ^ 1);
#else
      if (w_bf16) stageB_bf16(kt + 1, cur ^ 1); else stageB_f32(kt + 1, cur ^ 1);
#endif
    }

    v16bf afr[4], bfr[2];
    for (int mi = 0; mi < 4; ++mi) afr[mi] = ldfrag(&sA[cur][0], wm * 64 + mi * 16);
    for (int ni = 0; ni < 2; ++ni) bfr[ni] = ldfrag(&sB[cur][0], wn * 32 + ni * 16);

    for (int mi = 0; mi < 4; ++mi)
      for (int ni = 0; ni < 2; ++ni)
        acc[mi][ni] = __builtin_amdgcn_wmma_f32_16x16x32_bf16(
            false, afr[mi], false, bfr[ni], (short)0, acc[mi][ni], false, false);

    if (tdm_used && (kt + 1 < nk) && t < 32) wait_tensorcnt0();
    __syncthreads();
  }

  // ---- epilogue: C 16x16 layout (lanes 0-15: M=r, lanes 16-31: M=r+8) ----
  for (int mi = 0; mi < 4; ++mi)
    for (int ni = 0; ni < 2; ++ni) {
      int col = n0 + wn * 32 + ni * 16 + (lane & 15);
      int rb  = m0 + wm * 64 + mi * 16 + ((lane >> 4) << 3);
      for (int r = 0; r < 8; ++r) {
        float v = acc[mi][ni][r];
        int row = rb + r;
        if (addmask) v += addmask[(size_t)row * mask_ld + col];
        if (c_bf16) ((unsigned short*)Cp)[(size_t)row * ldc + col] = f2bf(v);
        else        ((float*)Cp)[(size_t)row * ldc + col] = v;
      }
    }
}

// =====================================================================
// Elementwise / normalization kernels (fp32 math, bandwidth-trivial)
// =====================================================================
__global__ __launch_bounds__(256)
void rms_bf16_kernel(const float* __restrict__ X, const float* __restrict__ W,
                     unsigned short* __restrict__ Y, int D)
{
  __shared__ float sred[256];
  int n = blockIdx.x, t = threadIdx.x;
  const float* x = X + (size_t)n * D;
  float ss = 0.f;
  for (int j = t; j < D; j += 256) { float v = x[j]; ss += v * v; }
  sred[t] = ss; __syncthreads();
  for (int s = 128; s > 0; s >>= 1) { if (t < s) sred[t] += sred[t + s]; __syncthreads(); }
  float inv = rsqrtf(sred[0] / (float)D + EPSV);
  for (int j = t; j < D; j += 256)
    Y[(size_t)n * D + j] = f2bf(x[j] * inv * (1.f + W[j]));
}

__global__ __launch_bounds__(256)
void residual_rms_kernel(const float* __restrict__ RES, const float* __restrict__ X,
                         const float* __restrict__ W, float* __restrict__ OUT,
                         const float* __restrict__ scale_ptr, int scale_idx, int D)
{
  __shared__ float sred[256];
  int n = blockIdx.x, t = threadIdx.x;
  const float* x = X + (size_t)n * D;
  float ss = 0.f;
  for (int j = t; j < D; j += 256) { float v = x[j]; ss += v * v; }
  sred[t] = ss; __syncthreads();
  for (int s = 128; s > 0; s >>= 1) { if (t < s) sred[t] += sred[t + s]; __syncthreads(); }
  float inv = rsqrtf(sred[0] / (float)D + EPSV);
  float sc  = scale_ptr ? scale_ptr[scale_idx] : 1.0f;
  for (int j = t; j < D; j += 256) {
    float r = RES[(size_t)n * D + j];
    OUT[(size_t)n * D + j] = (r + x[j] * inv * (1.f + W[j])) * sc;
  }
}

// q/k: per-(token,head) RMS norm with weight, then RoPE. bf16 head-major out;
// optional fp32 head-major out (for the K_i entries of d_out).
__global__ __launch_bounds__(256)
void qk_rope_kernel(const float* __restrict__ X, int NH,
                    const float* __restrict__ nw,
                    const float* __restrict__ cosT, const float* __restrict__ sinT,
                    unsigned short* __restrict__ Obf, float* __restrict__ Of32)
{
  __shared__ float sred[256];
  __shared__ float sn[256];
  int n = blockIdx.x, h = blockIdx.y, t = threadIdx.x;     // t == head-dim index
  float x = X[(size_t)n * NH * HDIM + (size_t)h * HDIM + t];
  sred[t] = x * x; __syncthreads();
  for (int s = 128; s > 0; s >>= 1) { if (t < s) sred[t] += sred[t + s]; __syncthreads(); }
  float inv = rsqrtf(sred[0] / (float)HDIM + EPSV);
  float rn  = x * inv * (1.f + nw[t]);
  sn[t] = rn; __syncthreads();
  float part = (t < HDIM / 2) ? -sn[t + HDIM / 2] : sn[t - HDIM / 2];
  float o = rn * cosT[(size_t)n * HDIM + t] + part * sinT[(size_t)n * HDIM + t];
  size_t oi = ((size_t)h * N_TOK + n) * HDIM + t;
  Obf[oi] = f2bf(o);
  if (Of32) Of32[oi] = o;
}

// v: per-(token,kv-head) L2-ish norm; bf16 transposed [KV][HD][N] for P@V GEMM,
// fp32 head-major [KV][N][HD] into d_out V_i.
__global__ __launch_bounds__(256)
void vnorm_kernel(const float* __restrict__ X,
                  unsigned short* __restrict__ Vt, float* __restrict__ Of32)
{
  __shared__ float sred[256];
  int n = blockIdx.x, h = blockIdx.y, t = threadIdx.x;
  float x = X[(size_t)n * NKV * HDIM + (size_t)h * HDIM + t];
  sred[t] = x * x; __syncthreads();
  for (int s = 128; s > 0; s >>= 1) { if (t < s) sred[t] += sred[t + s]; __syncthreads(); }
  float o = x * rsqrtf(sred[0] / (float)HDIM + EPSV);
  Vt[((size_t)h * HDIM + t) * N_TOK + n] = f2bf(o);        // [KV][HD][N]
  Of32[((size_t)h * N_TOK + n) * HDIM + t] = o;            // [KV][N][HD]
}

__global__ __launch_bounds__(256)
void softmax_kernel(const float* __restrict__ S, unsigned short* __restrict__ P)
{
  __shared__ float sred[256];
  int n = blockIdx.x, h = blockIdx.y, t = threadIdx.x;
  const float* row = S + ((size_t)h * N_TOK + n) * N_TOK;
  float v[4]; float mx = -3.0e38f;
  for (int j = 0; j < 4; ++j) { v[j] = row[t + 256 * j]; mx = fmaxf(mx, v[j]); }
  sred[t] = mx; __syncthreads();
  for (int s = 128; s > 0; s >>= 1) { if (t < s) sred[t] = fmaxf(sred[t], sred[t + s]); __syncthreads(); }
  mx = sred[0]; __syncthreads();
  float sum = 0.f;
  for (int j = 0; j < 4; ++j) { v[j] = __expf(v[j] - mx); sum += v[j]; }
  sred[t] = sum; __syncthreads();
  for (int s = 128; s > 0; s >>= 1) { if (t < s) sred[t] += sred[t + s]; __syncthreads(); }
  float inv = 1.f / sred[0];
  unsigned short* pr = P + ((size_t)h * N_TOK + n) * N_TOK;
  for (int j = 0; j < 4; ++j) pr[t + 256 * j] = f2bf(v[j] * inv);
}

__global__ void gelu_mul_kernel(const float* __restrict__ G, const float* __restrict__ U,
                                unsigned short* __restrict__ Y, long n)
{
  long i = (long)blockIdx.x * blockDim.x + threadIdx.x;
  if (i < n) Y[i] = f2bf(gelu_tanh(G[i]) * U[i]);
}

__global__ void gelu_gate_kernel(const float* __restrict__ PG, const float* __restrict__ PLC,
                                 int li, unsigned short* __restrict__ Y)
{
  long i = (long)blockIdx.x * blockDim.x + threadIdx.x;   // over N_TOK*HPLDIM
  long n = i / HPLDIM, j = i % HPLDIM;
  float sl = PLC[n * (long)(TOTLAYERS * HPLDIM) + (long)li * HPLDIM + j];
  Y[i] = f2bf(gelu_tanh(PG[i]) * sl);
}

__global__ void to_bf16_kernel(const float* __restrict__ X, unsigned short* __restrict__ Y, long n)
{
  long i = (long)blockIdx.x * blockDim.x + threadIdx.x;
  if (i < n) Y[i] = f2bf(X[i]);
}

// =====================================================================
// Host orchestration
// =====================================================================
static inline void launch_gemm(hipStream_t st, const unsigned short* A, int lda,
                               const void* W, int wbf, int ldw,
                               void* C, int cbf, int ldc,
                               const float* mask, int mld,
                               int M, int Nc, int K)
{
  dim3 g(Nc / BT_N, M / BT_M), b(256, 1, 1);
  gemm_wmma<<<g, b, 0, st>>>(A, lda, W, wbf, ldw, C, cbf, ldc, mask, mld, K, Nc);
}

extern "C" void kernel_launch(void* const* d_in, const int* in_sizes, int n_in,
                              void* d_out, int out_size, void* d_ws, size_t ws_size,
                              hipStream_t stream)
{
  const float* hidden = (const float*)d_in[0];
  const float* plc    = (const float*)d_in[1];
  const float* mask   = (const float*)d_in[2];
  const float* cos_s  = (const float*)d_in[3];
  const float* sin_s  = (const float*)d_in[4];
  const float* cos_f  = (const float*)d_in[5];
  const float* sin_f  = (const float*)d_in[6];
  const float* Wq   = (const float*)d_in[7];
  const float* Wk   = (const float*)d_in[8];
  const float* Wv   = (const float*)d_in[9];
  const float* Wo   = (const float*)d_in[10];
  const float* Wg   = (const float*)d_in[11];
  const float* Wu   = (const float*)d_in[12];
  const float* Wd   = (const float*)d_in[13];
  const float* Wplg = (const float*)d_in[14];
  const float* Wplp = (const float*)d_in[15];
  const float* ln_in = (const float*)d_in[16];
  const float* ln_pa = (const float*)d_in[17];
  const float* ln_pf = (const float*)d_in[18];
  const float* ln_ff = (const float*)d_in[19];
  const float* ln_pl = (const float*)d_in[20];
  const float* qn_w  = (const float*)d_in[21];
  const float* kn_w  = (const float*)d_in[22];
  const float* lscale = (const float*)d_in[23];
  float* out = (float*)d_out;

  // ---- workspace layout ----
  char* ws = (char*)d_ws;
  size_t off = 0;
  auto alloc = [&](size_t bytes) -> void* {
    void* p = ws + off;
    off += (bytes + 255) & ~(size_t)255;
    return p;
  };
  unsigned short* h_bf = (unsigned short*)alloc((size_t)N_TOK * DMODEL * 2);
  float* q_f = (float*)alloc((size_t)N_TOK * NHEAD * HDIM * 4);
  float* k_f = (float*)alloc((size_t)N_TOK * NKV * HDIM * 4);
  float* v_f = (float*)alloc((size_t)N_TOK * NKV * HDIM * 4);
  unsigned short* Qr = (unsigned short*)alloc((size_t)NHEAD * N_TOK * HDIM * 2);
  unsigned short* Kr = (unsigned short*)alloc((size_t)NKV * N_TOK * HDIM * 2);
  unsigned short* Vt = (unsigned short*)alloc((size_t)NKV * HDIM * N_TOK * 2);
  float* S  = (float*)alloc((size_t)NHEAD * N_TOK * N_TOK * 4);
  unsigned short* P = (unsigned short*)alloc((size_t)NHEAD * N_TOK * N_TOK * 2);
  unsigned short* AO = (unsigned short*)alloc((size_t)N_TOK * NHEAD * HDIM * 2);
  float* G  = (float*)alloc((size_t)N_TOK * FFDIM * 4);
  float* U  = (float*)alloc((size_t)N_TOK * FFDIM * 4);
  unsigned short* P2 = (unsigned short*)alloc((size_t)N_TOK * FFDIM * 2);
  float* pg = (float*)alloc((size_t)N_TOK * HPLDIM * 4);
  unsigned short* gated = (unsigned short*)alloc((size_t)N_TOK * HPLDIM * 2);
  float* proj = (float*)alloc((size_t)N_TOK * DMODEL * 4);
  float* hs   = (float*)alloc((size_t)N_TOK * DMODEL * 4);

  const size_t KVNH = (size_t)NKV * N_TOK * HDIM;
  const size_t OUT_KV_BASE = (size_t)N_TOK * DMODEL;

  hipMemcpyAsync(hs, hidden, (size_t)N_TOK * DMODEL * 4, hipMemcpyDeviceToDevice, stream);

  for (int i = 0; i < NLAYER; ++i) {
    const int li = STARTL + i;
    const bool full = ((li + 1) % 5 == 0);
    const float* cosT = full ? cos_f : cos_s;
    const float* sinT = full ? sin_f : sin_s;

    const float* Wq_i = Wq + (size_t)i * NHEAD * HDIM * DMODEL;
    const float* Wk_i = Wk + (size_t)i * NKV * HDIM * DMODEL;
    const float* Wv_i = Wv + (size_t)i * NKV * HDIM * DMODEL;
    const float* Wo_i = Wo + (size_t)i * DMODEL * NHEAD * HDIM;
    const float* Wg_i = Wg + (size_t)i * FFDIM * DMODEL;
    const float* Wu_i = Wu + (size_t)i * FFDIM * DMODEL;
    const float* Wd_i = Wd + (size_t)i * DMODEL * FFDIM;
    const float* Wplg_i = Wplg + (size_t)i * HPLDIM * DMODEL;
    const float* Wplp_i = Wplp + (size_t)i * DMODEL * HPLDIM;

    float* Kout = out + OUT_KV_BASE + (size_t)i * 2 * KVNH;
    float* Vout = Kout + KVNH;

    // ---- attention input norm + QKV projections ----
    rms_bf16_kernel<<<N_TOK, 256, 0, stream>>>(hs, ln_in + (size_t)i * DMODEL, h_bf, DMODEL);
    launch_gemm(stream, h_bf, DMODEL, Wq_i, 0, DMODEL, q_f, 0, NHEAD * HDIM,
                nullptr, 0, N_TOK, NHEAD * HDIM, DMODEL);
    launch_gemm(stream, h_bf, DMODEL, Wk_i, 0, DMODEL, k_f, 0, NKV * HDIM,
                nullptr, 0, N_TOK, NKV * HDIM, DMODEL);
    launch_gemm(stream, h_bf, DMODEL, Wv_i, 0, DMODEL, v_f, 0, NKV * HDIM,
                nullptr, 0, N_TOK, NKV * HDIM, DMODEL);

    // ---- q/k norm + rope; v norm (K/V fp32 straight into d_out) ----
    qk_rope_kernel<<<dim3(N_TOK, NHEAD), 256, 0, stream>>>(
        q_f, NHEAD, qn_w + (size_t)i * HDIM, cosT, sinT, Qr, nullptr);
    qk_rope_kernel<<<dim3(N_TOK, NKV), 256, 0, stream>>>(
        k_f, NKV, kn_w + (size_t)i * HDIM, cosT, sinT, Kr, Kout);
    vnorm_kernel<<<dim3(N_TOK, NKV), 256, 0, stream>>>(v_f, Vt, Vout);

    // ---- attention: S = Q K^T + mask (bf16 operands -> TDM staging path) ----
    for (int h = 0; h < NHEAD; ++h) {
      const unsigned short* Qh = Qr + (size_t)h * N_TOK * HDIM;
      const unsigned short* Kh = Kr + (size_t)(h >> 2) * N_TOK * HDIM;
      launch_gemm(stream, Qh, HDIM, (const void*)Kh, 1, HDIM,
                  S + (size_t)h * N_TOK * N_TOK, 0, N_TOK,
                  mask, N_TOK, N_TOK, N_TOK, HDIM);
    }
    softmax_kernel<<<dim3(N_TOK, NHEAD), 256, 0, stream>>>(S, P);
    // ---- O = P V  (writes interleaved [N, H*HD] bf16 directly) ----
    for (int h = 0; h < NHEAD; ++h) {
      const unsigned short* Ph  = P + (size_t)h * N_TOK * N_TOK;
      const unsigned short* Vth = Vt + (size_t)(h >> 2) * HDIM * N_TOK;
      launch_gemm(stream, Ph, N_TOK, (const void*)Vth, 1, N_TOK,
                  (void*)(AO + (size_t)h * HDIM), 1, NHEAD * HDIM,
                  nullptr, 0, N_TOK, HDIM, N_TOK);
    }
    // ---- output projection + residual ----
    launch_gemm(stream, AO, NHEAD * HDIM, Wo_i, 0, NHEAD * HDIM, proj, 0, DMODEL,
                nullptr, 0, N_TOK, DMODEL, NHEAD * HDIM);
    residual_rms_kernel<<<N_TOK, 256, 0, stream>>>(
        hs, proj, ln_pa + (size_t)i * DMODEL, hs, nullptr, 0, DMODEL);

    // ---- MLP ----
    rms_bf16_kernel<<<N_TOK, 256, 0, stream>>>(hs, ln_pf + (size_t)i * DMODEL, h_bf, DMODEL);
    launch_gemm(stream, h_bf, DMODEL, Wg_i, 0, DMODEL, G, 0, FFDIM,
                nullptr, 0, N_TOK, FFDIM, DMODEL);
    launch_gemm(stream, h_bf, DMODEL, Wu_i, 0, DMODEL, U, 0, FFDIM,
                nullptr, 0, N_TOK, FFDIM, DMODEL);
    gelu_mul_kernel<<<(int)(((long)N_TOK * FFDIM + 255) / 256), 256, 0, stream>>>(
        G, U, P2, (long)N_TOK * FFDIM);
    launch_gemm(stream, P2, FFDIM, Wd_i, 0, FFDIM, proj, 0, DMODEL,
                nullptr, 0, N_TOK, DMODEL, FFDIM);
    residual_rms_kernel<<<N_TOK, 256, 0, stream>>>(
        hs, proj, ln_ff + (size_t)i * DMODEL, hs, nullptr, 0, DMODEL);

    // ---- per-layer projection block ----
    to_bf16_kernel<<<(int)(((long)N_TOK * DMODEL + 255) / 256), 256, 0, stream>>>(
        hs, h_bf, (long)N_TOK * DMODEL);
    launch_gemm(stream, h_bf, DMODEL, Wplg_i, 0, DMODEL, pg, 0, HPLDIM,
                nullptr, 0, N_TOK, HPLDIM, DMODEL);
    gelu_gate_kernel<<<(int)(((long)N_TOK * HPLDIM + 255) / 256), 256, 0, stream>>>(
        pg, plc, li, gated);
    launch_gemm(stream, gated, HPLDIM, Wplp_i, 0, HPLDIM, proj, 0, DMODEL,
                nullptr, 0, N_TOK, DMODEL, HPLDIM);
    residual_rms_kernel<<<N_TOK, 256, 0, stream>>>(
        hs, proj, ln_pl + (size_t)i * DMODEL, hs, lscale, i, DMODEL);
  }

  hipMemcpyAsync(out, hs, (size_t)N_TOK * DMODEL * 4, hipMemcpyDeviceToDevice, stream);
}
